// FuzzyARTMAPClassifier_60026462929486
// MI455X (gfx1250) — matrix operand order repaired and checked by reading, not compiled, not run on hardware
//
#include <hip/hip_runtime.h>
#include <hip/hip_bf16.h>
#include <math.h>

// Problem constants (from reference)
#define B_SZ   128
#define C_MAX  1024
#define D_IN   1024
#define TWO_D  2048
#define NCLS   10
#define NPAD   16          // classes padded to 16 for WMMA N dim
#define ALPHA  1e-3f
#define GAMMA  1e-2f

// choice-kernel tiling
#define BM 16              // batch rows per workgroup (LDS tile: 16*2048*4 = 128 KB, fits 320KB WGP LDS)
#define BC 16              // categories per workgroup
#define CHOICE_THREADS 256 // 8 waves of 32

typedef __attribute__((ext_vector_type(16))) _Float16 v16h;
typedef __attribute__((ext_vector_type(8)))  _Float16 h8;
typedef __attribute__((ext_vector_type(8)))  float    v8f;

// ---------------------------------------------------------------------------
// ws layout (float offsets, all 64B-aligned):
//   [0..1]        : min bits, max bits (uint, monotone-comparable since x >= 0)
//   WS_CHOICE     : choice f32 [B_SZ * C_MAX]
//   WS_LSUM       : label_sums f32 [B_SZ * NPAD]
//   WS_CHOICE_H   : choice f16 [B_SZ * C_MAX]   (as _Float16*)
//   WS_BH         : transposed one-hot f16 [NPAD * C_MAX]
#define WS_CHOICE    16
#define WS_LSUM      (WS_CHOICE + B_SZ * C_MAX)                 // 131088
#define WS_CHOICE_H  (WS_LSUM + B_SZ * NPAD)                    // 133136
#define WS_BH        (WS_CHOICE_H + (B_SZ * C_MAX) / 2)         // 198672

// ---------------------------------------------------------------------------
__global__ void init_ws_kernel(unsigned int* mm) {
    if (threadIdx.x == 0) {
        mm[0] = 0x7F800000u;   // +inf  (atomicMin target)
        mm[1] = 0x00000000u;   // 0.0f  (atomicMax target; x >= 0)
    }
}

// Global min/max of x via bit-monotone unsigned atomics (valid: x in [0,1)).
__global__ void minmax_kernel(const float* __restrict__ x, unsigned int* __restrict__ mm, int n) {
    int i = blockIdx.x * blockDim.x + threadIdx.x;
    int stride = gridDim.x * blockDim.x;
    float lmin = 3.4e38f, lmax = 0.0f;
    for (; i < n; i += stride) {
        float v = x[i];
        lmin = fminf(lmin, v);
        lmax = fmaxf(lmax, v);
    }
    #pragma unroll
    for (int off = 16; off > 0; off >>= 1) {
        lmin = fminf(lmin, __shfl_xor(lmin, off, 32));
        lmax = fmaxf(lmax, __shfl_xor(lmax, off, 32));
    }
    if ((threadIdx.x & 31) == 0) {
        atomicMin(&mm[0], __float_as_uint(lmin));
        atomicMax(&mm[1], __float_as_uint(lmax));
    }
}

// Transposed one-hot with committed mask folded in (branchless):
// Bh[n*1024 + k] = (committed[k] && labels[k] == n) ? 1 : 0   (f16)
__global__ void build_onehot_kernel(const int* __restrict__ labels,
                                    const unsigned char* __restrict__ committed,
                                    _Float16* __restrict__ Bh) {
    int idx = blockIdx.x * blockDim.x + threadIdx.x;
    if (idx < NPAD * C_MAX) {
        int n  = idx >> 10;
        int k  = idx & (C_MAX - 1);
        int cm = committed[k];          // unconditional loads -> no divergent exec
        int lb = labels[k];
        Bh[idx] = (_Float16)(((cm != 0) & (lb == n)) ? 1.0f : 0.0f);
    }
}

// ---------------------------------------------------------------------------
// choice[b][c] = sum_k min(coded[b][k], T[c][k]) / (alpha + sum_k T[c][k] + gamma*counts[c])
// Tile: BM batch rows in LDS (128KB) x BC categories per workgroup.
// LDS fill uses CDNA5 async global->LDS copies (ASYNCcnt), then an in-place
// normalize/complement expansion. Template rows stream from L2 (8MB resident),
// each template element reused BM=16x from registers. ds_load pattern is
// bank-conflict-free (lane -> consecutive banks).
__global__ void choice_kernel(const float* __restrict__ x,
                              const float* __restrict__ T,
                              const int*   __restrict__ counts,
                              const unsigned int* __restrict__ mm,
                              float* __restrict__ choice,
                              _Float16* __restrict__ choice_h) {
    extern __shared__ float lds[];                 // [BM][TWO_D]
    const int cBase = blockIdx.x * BC;
    const int bBase = blockIdx.y * BM;

    const float mn  = __uint_as_float(mm[0]);
    const float mx  = __uint_as_float(mm[1]);
    const float inv = 1.0f / (mx - mn + 1e-10f);

    // Phase 1: async-copy raw x tile (BM x 1024 f32) into the low half of each
    // LDS row slot. b128 chunks; LDS byte address taken from vdst (dynamic LDS starts at 0).
    for (int ci = threadIdx.x; ci < (BM * D_IN) / 4; ci += blockDim.x) {
        int f0 = ci * 4;                                   // flat float idx in raw tile
        unsigned int ldsb = (unsigned int)(4 * f0 + ((f0 >> 10) << 12));  // skip complement half per row
        const float* g = x + (size_t)bBase * D_IN + f0;
        asm volatile("global_load_async_to_lds_b128 %0, %1, off"
                     :: "v"(ldsb), "v"(g) : "memory");
    }
    asm volatile("s_wait_asynccnt 0x0" ::: "memory");
    __syncthreads();

    // Phase 2: in-place normalize + complement-code expansion.
    for (int idx = threadIdx.x; idx < BM * D_IN; idx += blockDim.x) {
        int r = idx >> 10;
        int k = idx & (D_IN - 1);
        float v  = lds[r * TWO_D + k];
        float xn = (v - mn) * inv;
        lds[r * TWO_D + k]        = xn;
        lds[r * TWO_D + D_IN + k] = 1.0f - xn;
    }
    __syncthreads();

    const int wave = threadIdx.x >> 5;
    const int lane = threadIdx.x & 31;

    for (int cc = wave; cc < BC; cc += (CHOICE_THREADS / 32)) {
        const int c = cBase + cc;
        const float* __restrict__ trow = T + (size_t)c * TWO_D;

        float acc[BM];
        #pragma unroll
        for (int r = 0; r < BM; ++r) acc[r] = 0.0f;
        float tsum = 0.0f;

        for (int k = lane; k < TWO_D; k += 32) {
            __builtin_prefetch(trow + k + 256, 0, 0);   // global_prefetch_b8 (speculative, OOB-safe)
            float t = trow[k];
            tsum += t;
            #pragma unroll
            for (int r = 0; r < BM; ++r)
                acc[r] += fminf(lds[r * TWO_D + k], t);
        }

        // wave32 butterfly reductions
        #pragma unroll
        for (int off = 16; off > 0; off >>= 1) {
            tsum += __shfl_xor(tsum, off, 32);
            #pragma unroll
            for (int r = 0; r < BM; ++r)
                acc[r] += __shfl_xor(acc[r], off, 32);
        }

        if (lane == 0) {
            float denom = ALPHA + tsum + GAMMA * (float)counts[c];
            #pragma unroll
            for (int r = 0; r < BM; ++r) {
                float ch = acc[r] / denom;
                choice  [(size_t)(bBase + r) * C_MAX + c] = ch;
                choice_h[(size_t)(bBase + r) * C_MAX + c] = (_Float16)ch;
            }
        }
    }
}

// ---------------------------------------------------------------------------
// label_sums = choice @ Bmat via v_wmma_f32_16x16x32_f16, one wave per 16-row
// M tile. A fragments: two contiguous 16B h8 loads; B fragments: one contiguous
// 32B v16h load from prebuilt transposed one-hot. Layouts (A/B: contiguous k, not ISA order; C/D per ISA):
//   A (16-bit, 16x32): lanes 0-15 M=lane, halves 0-7 = K 0..7, 8-15 = K 8..15;
//                      lanes 16-31 same M, K bases 16 and 24.
//   B (16-bit, 32x16): lanes 0-15 N=lane hold K 0..15; lanes 16-31 hold K 16..31.
//   C/D (f32 16x16):   lanes 0-15 N=lane, vgpr v -> M=v; lanes 16-31 -> M=v+8.
__global__ void labelsum_wmma_kernel(const _Float16* __restrict__ choice_h,
                                     const _Float16* __restrict__ Bh,
                                     float* __restrict__ lsum /* [B_SZ][NPAD] */) {
    const int lane = threadIdx.x;          // 0..31, one wave per block
    const int m0   = blockIdx.x * 16;      // M tile base
    const int mrow = m0 + (lane & 15);
    const int n    = lane & 15;
    const int kA1  = (lane < 16) ? 0 : 16;
    const int kA2  = (lane < 16) ? 8 : 24;
    const int kB   = (lane < 16) ? 0 : 16;

    const _Float16* __restrict__ arow = choice_h + (size_t)mrow * C_MAX;
    const _Float16* __restrict__ brow = Bh + (size_t)n * C_MAX + kB;

    v8f c = {};
    for (int k0 = 0; k0 < C_MAX; k0 += 32) {
        h8 alo = *(const h8*)(arow + k0 + kA1);
        h8 ahi = *(const h8*)(arow + k0 + kA2);
        v16h a = __builtin_shufflevector(alo, ahi,
                 0, 1, 2, 3, 4, 5, 6, 7, 8, 9, 10, 11, 12, 13, 14, 15);
        v16h b = *(const v16h*)(brow + k0);
        c = __builtin_amdgcn_wmma_f32_16x16x32_f16(
                /*neg_a=*/false, a, /*neg_b=*/false, b,
                /*c_mod=*/(short)0, c, /*reuse_a=*/false, /*reuse_b=*/false);
    }

    const int mbase = m0 + ((lane < 16) ? 0 : 8);
    #pragma unroll
    for (int v = 0; v < 8; ++v)
        lsum[(mbase + v) * NPAD + n] = c[v];
}

// ---------------------------------------------------------------------------
// Per-row masked argmax (lowest-index tie-break, matching jnp.argmax) + one-hot select.
__global__ void finalize_kernel(const float* __restrict__ choice,
                                const unsigned char* __restrict__ committed,
                                const int*   __restrict__ labels,
                                const float* __restrict__ lsum,
                                float* __restrict__ out /* [B_SZ][NCLS] */) {
    __shared__ float smax[256];
    __shared__ int   sidx[256];
    const int b   = blockIdx.x;
    const int tid = threadIdx.x;
    const float* row = choice + (size_t)b * C_MAX;

    float bm = -INFINITY;
    int   bi = C_MAX;
    for (int cidx = tid; cidx < C_MAX; cidx += blockDim.x) {
        float v = committed[cidx] ? row[cidx] : -INFINITY;
        if (v > bm || (v == bm && cidx < bi)) { bm = v; bi = cidx; }
    }
    smax[tid] = bm; sidx[tid] = bi;
    __syncthreads();
    for (int s = 128; s > 0; s >>= 1) {
        if (tid < s) {
            float ov = smax[tid + s]; int oi = sidx[tid + s];
            if (ov > smax[tid] || (ov == smax[tid] && oi < sidx[tid])) {
                smax[tid] = ov; sidx[tid] = oi;
            }
        }
        __syncthreads();
    }
    if (tid < NCLS) {
        int pred = labels[sidx[0]];
        out[b * NCLS + tid] = (tid == pred) ? lsum[b * NPAD + tid] : 0.0f;
    }
}

// ---------------------------------------------------------------------------
extern "C" void kernel_launch(void* const* d_in, const int* in_sizes, int n_in,
                              void* d_out, int out_size, void* d_ws, size_t ws_size,
                              hipStream_t stream) {
    const float*         x         = (const float*)d_in[0];          // [128,1024]
    const float*         templates = (const float*)d_in[1];          // [1024,2048]
    const unsigned char* committed = (const unsigned char*)d_in[2];  // [1024] bool
    const int*           labels    = (const int*)d_in[3];            // [1024]
    const int*           counts    = (const int*)d_in[4];            // [1024]
    float*               out       = (float*)d_out;                  // [128,10]

    float*        ws       = (float*)d_ws;
    unsigned int* mm       = (unsigned int*)d_ws;
    float*        choice   = ws + WS_CHOICE;                     // [128,1024] f32
    float*        lsum     = ws + WS_LSUM;                       // [128,16]   f32
    _Float16*     choice_h = (_Float16*)(ws + WS_CHOICE_H);      // [128,1024] f16
    _Float16*     Bh       = (_Float16*)(ws + WS_BH);            // [16,1024]  f16

    // 1) init min/max slots
    init_ws_kernel<<<1, 32, 0, stream>>>(mm);

    // 2) global min/max of x
    minmax_kernel<<<128, 256, 0, stream>>>(x, mm, B_SZ * D_IN);

    // 3) transposed one-hot (committed folded in)
    build_onehot_kernel<<<(NPAD * C_MAX) / 256, 256, 0, stream>>>(labels, committed, Bh);

    // 4) fuzzy-AND choice: grid (C/BC, B/BM), 128KB dynamic LDS coded tile
    {
        dim3 grid(C_MAX / BC, B_SZ / BM);
        size_t shmem = (size_t)BM * TWO_D * sizeof(float);   // 131072 B, within 320KB WGP LDS
        choice_kernel<<<grid, CHOICE_THREADS, shmem, stream>>>(x, templates, counts, mm,
                                                               choice, choice_h);
    }

    // 5) label sums via WMMA: 8 M-tiles, one wave each
    labelsum_wmma_kernel<<<B_SZ / 16, 32, 0, stream>>>(choice_h, Bh, lsum);

    // 6) argmax + one-hot logits
    finalize_kernel<<<B_SZ, 256, 0, stream>>>(choice, committed, labels, lsum, out);
}
